// VectorQuantizer_25374666785027
// MI455X (gfx1250) — matrix operand order repaired and checked by reading, not compiled, not run on hardware
//
#include <hip/hip_runtime.h>

typedef __attribute__((ext_vector_type(16))) __bf16 v16bf;
typedef __attribute__((ext_vector_type(8)))  __bf16 v8bf;
typedef __attribute__((ext_vector_type(8)))  float  v8f;

static constexpr int CB_N   = 512;      // codebook entries
static constexpr int CB_H   = 256;      // entries per LDS pass
static constexpr int CDIM   = 64;       // channels
static constexpr int HWD    = 32768;    // H*W*D
static constexpr int TILE   = 128;      // points per block
static constexpr int ROWP   = 72;       // padded row stride in bf16 (144B, 16B-aligned)
static constexpr int NPTS   = 262144;   // B*H*W*D
static constexpr int NBLK   = NPTS / TILE;                  // 2048
static constexpr long long NELEM = (long long)NPTS * CDIM;  // 16777216
static constexpr float LOSS_SCALE = 1.25f / 16777216.0f;

// d_ws layout (bytes): [0,8192) partials | [8192, 8192+73728) bf16 codebook image | then 512 norms
static constexpr int WS_CB_OFF  = 8192;
static constexpr int WS_NRM_OFF = 8192 + CB_N * ROWP * 2;   // 81920

// ---- one-time codebook prep: fp32 -> padded bf16 image + (-0.5*||c||^2) norms ----
__global__ __launch_bounds__(256) void vq_prep(const float* __restrict__ cb,
                                               __bf16* __restrict__ cbbf,
                                               float* __restrict__ nrmg)
{
    const int t = threadIdx.x;
    for (int e = t; e < CB_N; e += 256) {
        const float* row = cb + e * CDIM;
        float s = 0.f;
        #pragma unroll
        for (int c = 0; c < CDIM; ++c) {
            const float v = row[c];
            s += v * v;
            cbbf[e * ROWP + c] = (__bf16)v;
        }
        nrmg[e] = -0.5f * s;
    }
}

__global__ __launch_bounds__(256) void vq_main(const float* __restrict__ x,
                                               const float* __restrict__ cb,
                                               const __bf16* __restrict__ cbbf,
                                               const float* __restrict__ nrmg,
                                               float* __restrict__ out,
                                               float* __restrict__ partial)
{
    __shared__ __align__(16) __bf16 cb_s[CB_H * ROWP];  // 36864 B
    __shared__ __align__(16) __bf16 xs[TILE * ROWP];    // 18432 B
    __shared__ float nrm_s[CB_N];                       //  2048 B
    __shared__ int   idx_s[TILE];                       //   512 B
    __shared__ float wred[8];

    const int t    = threadIdx.x;
    const int lane = t & 31;
    const int wav  = t >> 5;
    const int wg   = blockIdx.x;

    const int n0 = wg * TILE;          // tile never crosses a batch boundary (128 | 32768)
    const int b  = n0 / HWD;
    const int r0 = n0 % HWD;

    // ---- stage x tile into LDS as bf16 (float4 per lane: 512B per wave request) ----
    {
        const int pq = t & 31;          // group of 4 consecutive points
        const int c0 = t >> 5;          // 0..7
        const int gp = r0 + pq * 4;     // 16B-aligned (r0 % 128 == 0)
        #pragma unroll
        for (int i = 0; i < 8; ++i) {
            const int c = c0 + i * 8;
            const float4 v = *(const float4*)&x[(size_t)(b * CDIM + c) * HWD + gp];
            xs[(pq * 4 + 0) * ROWP + c] = (__bf16)v.x;
            xs[(pq * 4 + 1) * ROWP + c] = (__bf16)v.y;
            xs[(pq * 4 + 2) * ROWP + c] = (__bf16)v.z;
            xs[(pq * 4 + 3) * ROWP + c] = (__bf16)v.w;
        }
    }
    // ---- stage norms (once) ----
    for (int e = t; e < CB_N; e += 256) nrm_s[e] = nrmg[e];

    const int m  = lane & 15;
    const int hi = lane >> 4;
    v16bf a0, a1;

    float best[8];
    int   bidx[8];
    #pragma unroll
    for (int v = 0; v < 8; ++v) { best[v] = -3.0e38f; bidx[v] = 0; }

    for (int half = 0; half < 2; ++half) {
        // ---- copy prebuilt bf16 codebook half image into LDS (b128 both sides) ----
        {
            const uint4* src = (const uint4*)(cbbf + half * CB_H * ROWP);
            uint4*       dst = (uint4*)cb_s;
            #pragma unroll
            for (int k = 0; k < (CB_H * ROWP * 2 / 16) / 256; ++k)   // 9 per thread
                dst[t + k * 256] = src[t + k * 256];
        }
        __syncthreads();

        if (half == 0) {
            // A fragments (16-bit A 16x32 layout: lane m=row; K chunks kb..kb+7, kb+16..kb+23)
            const __bf16* xrow = &xs[(wav * 16 + m) * ROWP];
            const int kb = hi * 8;
            v8bf p0 = *(const v8bf*)(xrow + kb);
            v8bf p1 = *(const v8bf*)(xrow + 16 + kb);
            a0 = __builtin_shufflevector(p0, p1, 0,1,2,3,4,5,6,7,8,9,10,11,12,13,14,15);
            v8bf p2 = *(const v8bf*)(xrow + 32 + kb);
            v8bf p3 = *(const v8bf*)(xrow + 48 + kb);
            a1 = __builtin_shufflevector(p2, p3, 0,1,2,3,4,5,6,7,8,9,10,11,12,13,14,15);
        }

        #pragma unroll 2
        for (int jt = 0; jt < CB_H / 16; ++jt) {
            const int el = jt * 16 + m;            // local entry = B column N
            const int eg = half * CB_H + el;
            // B 32x16 layout: same K chunks as A (kb..kb+7, kb+16..kb+23)
            const __bf16* crow = &cb_s[el * ROWP + hi * 8];
            v8bf q0 = *(const v8bf*)(crow);
            v8bf q1 = *(const v8bf*)(crow + 16);
            v16bf b0 = __builtin_shufflevector(q0, q1, 0,1,2,3,4,5,6,7,8,9,10,11,12,13,14,15);
            v8bf q2 = *(const v8bf*)(crow + 32);
            v8bf q3 = *(const v8bf*)(crow + 48);
            v16bf b1 = __builtin_shufflevector(q2, q3, 0,1,2,3,4,5,6,7,8,9,10,11,12,13,14,15);

            // bias accumulator with -0.5*||c||^2 BEFORE the WMMAs (off the critical path)
            const float nb = nrm_s[eg];
            v8f acc;
            #pragma unroll
            for (int v = 0; v < 8; ++v) acc[v] = nb;
            acc = __builtin_amdgcn_wmma_f32_16x16x32_bf16(false, a0, false, b0, (short)0, acc, false, false);
            acc = __builtin_amdgcn_wmma_f32_16x16x32_bf16(false, a1, false, b1, (short)0, acc, false, false);

            #pragma unroll
            for (int v = 0; v < 8; ++v) {          // argmin dist == argmax (x.c - 0.5||c||^2)
                if (acc[v] > best[v]) { best[v] = acc[v]; bidx[v] = eg; }
            }
        }
        __syncthreads();   // protect cb_s before next pass overwrites it
    }

    // ---- reduce over the 16 code columns (xor 1,2,4,8 stays within each 16-lane half) ----
    #pragma unroll
    for (int off = 1; off < 16; off <<= 1) {
        #pragma unroll
        for (int v = 0; v < 8; ++v) {
            const float os = __shfl_xor(best[v], off, 32);
            const int   oi = __shfl_xor(bidx[v], off, 32);
            if (os > best[v] || (os == best[v] && oi < bidx[v])) { best[v] = os; bidx[v] = oi; }
        }
    }
    // C/D layout: lanes 0-15 hold rows M=v, lanes 16-31 rows M=v+8
    if (m == 0) {
        #pragma unroll
        for (int v = 0; v < 8; ++v) idx_s[wav * 16 + hi * 8 + v] = bidx[v];
    }
    __syncthreads();

    // ---- gather fp32 codebook rows, write quant (b128 stores), accumulate loss ----
    float lsum = 0.f;
    {
        const int pq = t & 31;
        const int c0 = t >> 5;
        const int gp = r0 + pq * 4;
        const int p4 = pq * 4;
        const int e0 = idx_s[p4 + 0] * CDIM;
        const int e1 = idx_s[p4 + 1] * CDIM;
        const int e2 = idx_s[p4 + 2] * CDIM;
        const int e3 = idx_s[p4 + 3] * CDIM;
        #pragma unroll
        for (int i = 0; i < 8; ++i) {
            const int c = c0 + i * 8;
            const size_t g = (size_t)(b * CDIM + c) * HWD + gp;
            const float4 xv = *(const float4*)&x[g];
            float4 q;
            q.x = cb[e0 + c];
            q.y = cb[e1 + c];
            q.z = cb[e2 + c];
            q.w = cb[e3 + c];
            const float d0 = q.x - xv.x, d1 = q.y - xv.y, d2 = q.z - xv.z, d3 = q.w - xv.w;
            lsum += d0 * d0 + d1 * d1 + d2 * d2 + d3 * d3;
            *(float4*)&out[g] = q;
        }
    }
    #pragma unroll
    for (int off = 16; off >= 1; off >>= 1) lsum += __shfl_xor(lsum, off, 32);
    if (lane == 0) wred[wav] = lsum;
    __syncthreads();
    if (t == 0) {
        float s = 0.f;
        #pragma unroll
        for (int w = 0; w < 8; ++w) s += wred[w];
        partial[wg] = s;
    }
}

__global__ __launch_bounds__(256) void vq_loss_reduce(const float* __restrict__ partial,
                                                      float* __restrict__ out_loss)
{
    __shared__ float wred[8];
    const int t = threadIdx.x;
    float s = 0.f;
    #pragma unroll
    for (int k = 0; k < NBLK / 256; ++k) s += partial[t + k * 256];
    #pragma unroll
    for (int off = 16; off >= 1; off >>= 1) s += __shfl_xor(s, off, 32);
    if ((t & 31) == 0) wred[t >> 5] = s;
    __syncthreads();
    if (t == 0) {
        float tot = 0.f;
        #pragma unroll
        for (int i = 0; i < 8; ++i) tot += wred[i];
        *out_loss = LOSS_SCALE * tot;
    }
}

extern "C" void kernel_launch(void* const* d_in, const int* in_sizes, int n_in,
                              void* d_out, int out_size, void* d_ws, size_t ws_size,
                              hipStream_t stream) {
    const float* x  = (const float*)d_in[0];   // [8,64,32,32,32] fp32
    const float* cb = (const float*)d_in[1];   // [512,64] fp32
    float* out      = (float*)d_out;           // 16777216 quant + 1 loss

    float*  partial = (float*)d_ws;
    __bf16* cbbf    = (__bf16*)((char*)d_ws + WS_CB_OFF);
    float*  nrmg    = (float*)((char*)d_ws + WS_NRM_OFF);

    vq_prep<<<1, 256, 0, stream>>>(cb, cbbf, nrmg);
    vq_main<<<NBLK, 256, 0, stream>>>(x, cb, cbbf, nrmg, out, partial);
    vq_loss_reduce<<<1, 256, 0, stream>>>(partial, out + NELEM);
}
